// DGCNN_MODULAR_21698174780227
// MI455X (gfx1250) — matrix operand order repaired and checked by reading, not compiled, not run on hardware
//
#include <hip/hip_runtime.h>
#include <hip/hip_bf16.h>

#define BB   4
#define NN   2048
#define KNB  20
#define ACTW 960

typedef __attribute__((ext_vector_type(16))) _Float16 v16h;
typedef __attribute__((ext_vector_type(8)))  float    v8f;

__device__ __forceinline__ v8f wmma16(v16h a, v16h b, v8f c) {
  // D = A(16x32 f16) * B(32x16 f16) + C(16x16 f32)
  return __builtin_amdgcn_wmma_f32_16x16x32_f16(false, a, false, b, (short)0, c, false, false);
}

__device__ __forceinline__ float leaky(float x) { return x >= 0.f ? x : 0.2f * x; }

// ---------------------------------------------------------------- utilities
__global__ void k_zero(float* p, int n) {
  for (int i = blockIdx.x * blockDim.x + threadIdx.x; i < n; i += gridDim.x * blockDim.x)
    p[i] = 0.f;
}

__global__ void k_x_to_h(const float* __restrict__ x, _Float16* __restrict__ xh) {
  size_t total = (size_t)BB * NN * 4;
  for (size_t i = blockIdx.x * (size_t)blockDim.x + threadIdx.x; i < total;
       i += (size_t)gridDim.x * blockDim.x) {
    int c = (int)(i & 3);
    size_t bn = i >> 2;
    xh[i] = (c < 3) ? (_Float16)x[bn * 3 + c] : (_Float16)0.f;
  }
}

__global__ void k_convert_w(const float* __restrict__ W, _Float16* __restrict__ Wh,
                            int O, int Cin, int stride) {
  size_t total = (size_t)O * stride;
  for (size_t i = blockIdx.x * (size_t)blockDim.x + threadIdx.x; i < total;
       i += (size_t)gridDim.x * blockDim.x) {
    int c = (int)(i % stride);
    size_t o = i / stride;
    Wh[i] = (c < Cin) ? (_Float16)W[o * Cin + c] : (_Float16)0.f;
  }
}

// ------------------------------------------------- edge-conv GEMM (+stats, +max over K)
// rows = (n,k) pairs (M-tile = 80 = 4 points * 20 neighbors = 5 wmma row tiles)
// cols = 64 output channels (4 waves * 16)
__global__ void __launch_bounds__(128)
k_edge_gemm(const _Float16* __restrict__ in, int istride, int Cin,
            const int* __restrict__ nbr,
            const _Float16* __restrict__ Wh, int wstride,
            int O, float* __restrict__ z,
            float* __restrict__ gsum, float* __restrict__ gsumsq)
{
  __shared__ _Float16 As[80][32];
  __shared__ float    Yd[4][80][16];
  __shared__ float    Red[4][32][2];

  const int tid  = threadIdx.x;
  const int lane = tid & 31;
  const int w    = tid >> 5;
  const int h    = lane >> 4;
  const int col  = lane & 15;
  const int b    = blockIdx.z;
  const int n0   = blockIdx.x * 4;
  const int oc0  = blockIdx.y * 64 + w * 16;
  const int Cin2 = 2 * Cin;

  v8f acc[5] = {};

  for (int kc = 0; kc < wstride; kc += 32) {
    __syncthreads();
    // build edge-feature tile: e = [ o[nb]-o[n] ; o[n] ] (zero-padded)
    for (int i = tid; i < 80 * 32; i += 128) {
      int m = i >> 5, cc = i & 31;
      int c = kc + cc;
      int n = n0 + m / 20;
      int k = m % 20;
      _Float16 v = (_Float16)0.f;
      if (c < Cin2) {
        size_t rowc = ((size_t)b * NN + n) * istride;
        if (c < Cin) {
          int nb = nbr[((size_t)b * NN + n) * KNB + k];
          v = in[((size_t)b * NN + nb) * istride + c] - in[rowc + c];
        } else {
          v = in[rowc + (c - Cin)];
        }
      }
      As[m][cc] = v;
    }
    __syncthreads();

    v16h bf;
    {
      const _Float16* pb = Wh + (size_t)(oc0 + col) * wstride + kc + h * 16;
      #pragma unroll
      for (int j = 0; j < 16; ++j) bf[j] = pb[j];
    }
    #pragma unroll
    for (int r = 0; r < 5; ++r) {
      v16h af;
      int m = r * 16 + col;
      #pragma unroll
      for (int j = 0; j < 8; ++j) {
        int kb = j * 2 + h * 16;
        af[2 * j]     = As[m][kb];
        af[2 * j + 1] = As[m][kb + 1];
      }
      acc[r] = wmma16(af, bf, acc[r]);
    }
  }

  // dump tile, accumulate per-channel sums, reduce max over K
  __syncthreads();
  #pragma unroll
  for (int r = 0; r < 5; ++r)
    #pragma unroll
    for (int v = 0; v < 8; ++v)
      Yd[w][r * 16 + v + 8 * h][col] = acc[r][v];
  __syncthreads();

  float s1 = 0.f, s2 = 0.f;
  for (int rr = 0; rr < 40; ++rr) {
    float yv = Yd[w][40 * h + rr][col];
    s1 += yv; s2 += yv * yv;
  }
  for (int t = 0; t < 2; ++t) {
    int nl = 2 * h + t;
    float mx = -3.4e38f;
    for (int kk = 0; kk < 20; ++kk) mx = fmaxf(mx, Yd[w][nl * 20 + kk][col]);
    z[((size_t)b * NN + (n0 + nl)) * O + oc0 + col] = mx;
  }
  Red[w][lane][0] = s1;
  Red[w][lane][1] = s2;
  __syncthreads();
  if (h == 0) {
    atomicAdd(&gsum[oc0 + col],   Red[w][lane][0] + Red[w][lane + 16][0]);
    atomicAdd(&gsumsq[oc0 + col], Red[w][lane][1] + Red[w][lane + 16][1]);
  }
}

// ------------------------------------------------- Gram matrix S = O * O^T (per batch)
__global__ void __launch_bounds__(128)
k_gram(const _Float16* __restrict__ in, int istride, int Cin,
       float* __restrict__ S, int b)
{
  const int tid  = threadIdx.x;
  const int lane = tid & 31;
  const int w    = tid >> 5;
  const int h    = lane >> 4;
  const int col  = lane & 15;
  const int r0   = blockIdx.y * 64;
  const int c0   = blockIdx.x * 64 + w * 16;

  v8f acc[4] = {};
  for (int kc = 0; kc < Cin; kc += 32) {
    v16h bf;
    {
      const _Float16* pb = in + ((size_t)b * NN + (c0 + col)) * istride + kc + h * 16;
      #pragma unroll
      for (int j = 0; j < 16; ++j) bf[j] = pb[j];
    }
    #pragma unroll
    for (int r = 0; r < 4; ++r) {
      v16h af;
      const _Float16* pa = in + ((size_t)b * NN + (r0 + r * 16 + col)) * istride + kc;
      #pragma unroll
      for (int j = 0; j < 8; ++j) {
        int kb = j * 2 + h * 16;
        af[2 * j]     = pa[kb];
        af[2 * j + 1] = pa[kb + 1];
      }
      acc[r] = wmma16(af, bf, acc[r]);
    }
  }
  #pragma unroll
  for (int r = 0; r < 4; ++r)
    #pragma unroll
    for (int v = 0; v < 8; ++v)
      S[(size_t)(r0 + r * 16 + v + 8 * h) * NN + c0 + col] = acc[r][v];
}

__global__ void k_diag(const float* __restrict__ S, float* __restrict__ xx) {
  int i = blockIdx.x * blockDim.x + threadIdx.x;
  if (i < NN) xx[i] = S[(size_t)i * NN + i];
}

// per-row stable top-20 of neg_dist = 2*S - xx[n] - xx[m]
__global__ void k_topk(const float* __restrict__ S, const float* __restrict__ xx,
                       int* __restrict__ idxo, int b)
{
  int n = blockIdx.x * blockDim.x + threadIdx.x;
  if (n >= NN) return;
  float vals[KNB];
  int   inds[KNB];
  int cnt = 0;
  float xn = xx[n];
  const float* row = S + (size_t)n * NN;
  for (int m = 0; m < NN; ++m) {
    float nd = 2.f * row[m] - xn - xx[m];
    if (cnt == KNB && nd <= vals[KNB - 1]) continue;
    int p = (cnt < KNB) ? cnt : (KNB - 1);
    while (p > 0 && vals[p - 1] < nd) {
      vals[p] = vals[p - 1];
      inds[p] = inds[p - 1];
      --p;
    }
    vals[p] = nd;
    inds[p] = m;
    if (cnt < KNB) ++cnt;
  }
  for (int k = 0; k < KNB; ++k) idxo[((size_t)b * NN + n) * KNB + k] = inds[k];
}

// ------------------------------------------------- conv1d GEMM (+stats)
__global__ void __launch_bounds__(128)
k_lin_gemm(const _Float16* __restrict__ in, int istride, int Cin,
           const _Float16* __restrict__ Wh, int wstride, int O,
           float* __restrict__ y, float* __restrict__ gsum, float* __restrict__ gsumsq)
{
  __shared__ float Red[4][32][2];
  const int tid  = threadIdx.x;
  const int lane = tid & 31;
  const int w    = tid >> 5;
  const int h    = lane >> 4;
  const int col  = lane & 15;
  const int b    = blockIdx.z;
  const int n0   = blockIdx.x * 64;
  const int oc0  = blockIdx.y * 64 + w * 16;

  v8f acc[4] = {};
  for (int kc = 0; kc < Cin; kc += 32) {
    v16h bf;
    {
      const _Float16* pb = Wh + (size_t)(oc0 + col) * wstride + kc + h * 16;
      #pragma unroll
      for (int j = 0; j < 16; ++j) bf[j] = pb[j];
    }
    #pragma unroll
    for (int r = 0; r < 4; ++r) {
      v16h af;
      const _Float16* pa = in + ((size_t)b * NN + n0 + r * 16 + col) * istride + kc;
      #pragma unroll
      for (int j = 0; j < 8; ++j) {
        int kb = j * 2 + h * 16;
        af[2 * j]     = pa[kb];
        af[2 * j + 1] = pa[kb + 1];
      }
      acc[r] = wmma16(af, bf, acc[r]);
    }
  }
  float s1 = 0.f, s2 = 0.f;
  #pragma unroll
  for (int r = 0; r < 4; ++r)
    #pragma unroll
    for (int v = 0; v < 8; ++v) {
      float yv = acc[r][v];
      y[((size_t)b * NN + n0 + r * 16 + v + 8 * h) * O + oc0 + col] = yv;
      s1 += yv; s2 += yv * yv;
    }
  Red[w][lane][0] = s1;
  Red[w][lane][1] = s2;
  __syncthreads();
  if (h == 0) {
    atomicAdd(&gsum[oc0 + col],   Red[w][lane][0] + Red[w][lane + 16][0]);
    atomicAdd(&gsumsq[oc0 + col], Red[w][lane][1] + Red[w][lane + 16][1]);
  }
}

// ------------------------------------------------- BN finalize / apply
__global__ void k_finalize(const float* __restrict__ sum, const float* __restrict__ sumsq,
                           const float* __restrict__ g, const float* __restrict__ bb,
                           float cnt, float* __restrict__ scale, float* __restrict__ shift, int O)
{
  int o = blockIdx.x * blockDim.x + threadIdx.x;
  if (o >= O) return;
  float m = sum[o] / cnt;
  float v = sumsq[o] / cnt - m * m;
  float s = g[o] * rsqrtf(v + 1e-5f);
  scale[o] = s;
  shift[o] = bb[o] - m * s;
}

__global__ void k_apply_act(const float* __restrict__ z, const float* __restrict__ scale,
                            const float* __restrict__ shift, _Float16* __restrict__ act,
                            int coloff, int O)
{
  size_t total = (size_t)BB * NN * O;
  for (size_t i = blockIdx.x * (size_t)blockDim.x + threadIdx.x; i < total;
       i += (size_t)gridDim.x * blockDim.x) {
    int o = (int)(i % O);
    size_t bn = i / O;
    float v = leaky(z[i] * scale[o] + shift[o]);
    act[bn * ACTW + coloff + o] = (_Float16)v;
  }
}

__global__ void k_apply_feat(const float* __restrict__ y, const float* __restrict__ scale,
                             const float* __restrict__ shift, float* __restrict__ feat)
{
  size_t total = (size_t)BB * NN * 512;
  for (size_t i = blockIdx.x * (size_t)blockDim.x + threadIdx.x; i < total;
       i += (size_t)gridDim.x * blockDim.x) {
    int o = (int)(i % 512);
    size_t bn = i / 512;
    int n = (int)(bn % NN);
    int b = (int)(bn / NN);
    float v = leaky(y[i] * scale[o] + shift[o]);
    feat[((size_t)b * 512 + o) * NN + n] = v;
  }
}

// ------------------------------------------------- pooling + MLP head
__global__ void __launch_bounds__(256)
k_pool(const float* __restrict__ feat, float* __restrict__ a)
{
  int b = blockIdx.x / 512, o = blockIdx.x % 512;
  const float* p = feat + ((size_t)b * 512 + o) * NN;
  float mx = -3.4e38f, sm = 0.f;
  for (int n = threadIdx.x; n < NN; n += 256) {
    float v = p[n];
    mx = fmaxf(mx, v);
    sm += v;
  }
  __shared__ float smx[256], ssm[256];
  int tid = threadIdx.x;
  smx[tid] = mx; ssm[tid] = sm;
  __syncthreads();
  for (int s = 128; s > 0; s >>= 1) {
    if (tid < s) {
      smx[tid] = fmaxf(smx[tid], smx[tid + s]);
      ssm[tid] += ssm[tid + s];
    }
    __syncthreads();
  }
  if (tid == 0) {
    a[(size_t)b * 1024 + o]       = smx[0];
    a[(size_t)b * 1024 + 512 + o] = ssm[0] / (float)NN;
  }
}

__global__ void __launch_bounds__(256)
k_head(const float* __restrict__ a,
       const float* __restrict__ W1, const float* __restrict__ g6, const float* __restrict__ b6,
       const float* __restrict__ W2, const float* __restrict__ bb2,
       const float* __restrict__ g7, const float* __restrict__ b7,
       const float* __restrict__ W3, const float* __restrict__ bb3,
       float* __restrict__ out)
{
  __shared__ float h1[4][1024];
  __shared__ float h2[4][512];
  int tid = threadIdx.x;

  for (int t = tid; t < 4 * 1024; t += 256) {
    int i = t >> 10, o = t & 1023;
    const float* wr = W1 + (size_t)o * 1024;
    const float* ar = a + (size_t)i * 1024;
    float s = 0.f;
    for (int c = 0; c < 1024; ++c) s += ar[c] * wr[c];
    h1[i][o] = s;
  }
  __syncthreads();
  for (int o = tid; o < 1024; o += 256) {
    float m = 0.f;
    for (int i = 0; i < 4; ++i) m += h1[i][o];
    m *= 0.25f;
    float v = 0.f;
    for (int i = 0; i < 4; ++i) { float d = h1[i][o] - m; v += d * d; }
    v *= 0.25f;
    float sc = g6[o] * rsqrtf(v + 1e-5f);
    for (int i = 0; i < 4; ++i) h1[i][o] = leaky((h1[i][o] - m) * sc + b6[o]);
  }
  __syncthreads();
  for (int t = tid; t < 4 * 512; t += 256) {
    int i = t >> 9, o = t & 511;
    const float* wr = W2 + (size_t)o * 1024;
    float s = bb2[o];
    for (int c = 0; c < 1024; ++c) s += h1[i][c] * wr[c];
    h2[i][o] = s;
  }
  __syncthreads();
  for (int o = tid; o < 512; o += 256) {
    float m = 0.f;
    for (int i = 0; i < 4; ++i) m += h2[i][o];
    m *= 0.25f;
    float v = 0.f;
    for (int i = 0; i < 4; ++i) { float d = h2[i][o] - m; v += d * d; }
    v *= 0.25f;
    float sc = g7[o] * rsqrtf(v + 1e-5f);
    for (int i = 0; i < 4; ++i) h2[i][o] = leaky((h2[i][o] - m) * sc + b7[o]);
  }
  __syncthreads();
  for (int t = tid; t < 4 * 512; t += 256) {
    int i = t >> 9, o = t & 511;
    const float* wr = W3 + (size_t)o * 512;
    float s = bb3[o];
    for (int c = 0; c < 512; ++c) s += h2[i][c] * wr[c];
    out[(size_t)i * 512 + o] = s;
  }
}

// ---------------------------------------------------------------- launcher
extern "C" void kernel_launch(void* const* d_in, const int* in_sizes, int n_in,
                              void* d_out, int out_size, void* d_ws, size_t ws_size,
                              hipStream_t stream)
{
  (void)in_sizes; (void)n_in; (void)out_size; (void)ws_size;

  const float* x      = (const float*)d_in[0];
  const int*   sneigh = (const int*)d_in[1];
  const float* W0 = (const float*)d_in[2];
  const float* g0 = (const float*)d_in[3];
  const float* b0 = (const float*)d_in[4];
  const float* W1 = (const float*)d_in[5];
  const float* g1 = (const float*)d_in[6];
  const float* b1 = (const float*)d_in[7];
  const float* W2 = (const float*)d_in[8];
  const float* g2 = (const float*)d_in[9];
  const float* b2 = (const float*)d_in[10];
  const float* W3 = (const float*)d_in[11];
  const float* g3 = (const float*)d_in[12];
  const float* b3 = (const float*)d_in[13];
  const float* W4 = (const float*)d_in[14];
  const float* g4 = (const float*)d_in[15];
  const float* b4 = (const float*)d_in[16];
  const float* lin1_W = (const float*)d_in[17];
  const float* g6 = (const float*)d_in[18];
  const float* b6 = (const float*)d_in[19];
  const float* lin2_W = (const float*)d_in[20];
  const float* lin2_b = (const float*)d_in[21];
  const float* g7 = (const float*)d_in[22];
  const float* b7 = (const float*)d_in[23];
  const float* lin3_W = (const float*)d_in[24];
  const float* lin3_b = (const float*)d_in[25];

  float* out  = (float*)d_out;      // (B,512) = 2048 floats
  float* feat = out + 2048;         // (B,512,N)

  char* p = (char*)d_ws;
  auto alloc = [&](size_t bytes) -> char* {
    char* r = p;
    p += (bytes + 255) & ~(size_t)255;
    return r;
  };
  _Float16* xh   = (_Float16*)alloc((size_t)BB * NN * 4 * 2);
  _Float16* act  = (_Float16*)alloc((size_t)BB * NN * ACTW * 2);
  _Float16* w0h  = (_Float16*)alloc((size_t)64 * 32 * 2);
  _Float16* w1h  = (_Float16*)alloc((size_t)128 * 128 * 2);
  _Float16* w2h  = (_Float16*)alloc((size_t)256 * 256 * 2);
  _Float16* w3h  = (_Float16*)alloc((size_t)512 * 512 * 2);
  _Float16* w4h  = (_Float16*)alloc((size_t)512 * 960 * 2);
  int*      knn  = (int*)alloc((size_t)BB * NN * KNB * 4);
  float*    S    = (float*)alloc((size_t)NN * NN * 4);
  float*    z    = (float*)alloc((size_t)BB * NN * 512 * 4);
  float*    stats = (float*)alloc(1024 * 4);   // [sum | sumsq]
  float*    scale = (float*)alloc(512 * 4);
  float*    shift = (float*)alloc(512 * 4);
  float*    xx    = (float*)alloc(NN * 4);
  float*    abuf  = (float*)alloc((size_t)BB * 1024 * 4);

  const float cntK = (float)((size_t)BB * NN * KNB);
  const float cntN = (float)((size_t)BB * NN);

  k_x_to_h<<<128, 256, 0, stream>>>(x, xh);
  k_convert_w<<<(64 * 32 + 255) / 256, 256, 0, stream>>>(W0, w0h, 64, 6, 32);
  k_convert_w<<<(128 * 128 + 255) / 256, 256, 0, stream>>>(W1, w1h, 128, 128, 128);
  k_convert_w<<<(256 * 256 + 255) / 256, 256, 0, stream>>>(W2, w2h, 256, 256, 256);
  k_convert_w<<<(512 * 512 + 255) / 256, 256, 0, stream>>>(W3, w3h, 512, 512, 512);
  k_convert_w<<<(512 * 960 + 255) / 256, 256, 0, stream>>>(W4, w4h, 512, 960, 960);

  // ---- Layer 0: edge conv on raw points, given neighbors
  k_zero<<<1, 256, 0, stream>>>(stats, 2 * 64);
  k_edge_gemm<<<dim3(NN / 4, 1, BB), 128, 0, stream>>>(xh, 4, 3, sneigh, w0h, 32, 64,
                                                       z, stats, stats + 64);
  k_finalize<<<1, 64, 0, stream>>>(stats, stats + 64, g0, b0, cntK, scale, shift, 64);
  k_apply_act<<<2048, 256, 0, stream>>>(z, scale, shift, act, 0, 64);

  struct Layer { int coloff_in, Cin; const _Float16* Wh; int wstride, O; int coloff_out;
                 const float *g, *b; };
  Layer layers[3] = {
    { 0,   64,  w1h, 128, 128, 64,  g1, b1 },
    { 64,  128, w2h, 256, 256, 192, g2, b2 },
    { 192, 256, w3h, 512, 512, 448, g3, b3 },
  };
  for (int li = 0; li < 3; ++li) {
    Layer L = layers[li];
    // KNN via Gram matrix (stream-serialized per batch, S reused)
    for (int b = 0; b < BB; ++b) {
      k_gram<<<dim3(NN / 64, NN / 64), 128, 0, stream>>>(act + L.coloff_in, ACTW, L.Cin, S, b);
      k_diag<<<NN / 256, 256, 0, stream>>>(S, xx);
      k_topk<<<NN / 256, 256, 0, stream>>>(S, xx, knn, b);
    }
    k_zero<<<1, 256, 0, stream>>>(stats, 2 * L.O);
    k_edge_gemm<<<dim3(NN / 4, L.O / 64, BB), 128, 0, stream>>>(
        act + L.coloff_in, ACTW, L.Cin, knn, L.Wh, L.wstride, L.O, z, stats, stats + L.O);
    k_finalize<<<(L.O + 63) / 64, 64, 0, stream>>>(stats, stats + L.O, L.g, L.b, cntK,
                                                   scale, shift, L.O);
    int total = BB * NN * L.O;
    k_apply_act<<<(total + 255) / 256, 256, 0, stream>>>(z, scale, shift, act, L.coloff_out, L.O);
  }

  // ---- conv1d over concatenated features (960 -> 512)
  k_zero<<<1, 256, 0, stream>>>(stats, 2 * 512);
  k_lin_gemm<<<dim3(NN / 64, 512 / 64, BB), 128, 0, stream>>>(act, ACTW, 960, w4h, 960, 512,
                                                              z, stats, stats + 512);
  k_finalize<<<8, 64, 0, stream>>>(stats, stats + 512, g4, b4, cntN, scale, shift, 512);
  k_apply_feat<<<16384, 256, 0, stream>>>(z, scale, shift, feat);

  // ---- pooling + MLP head
  k_pool<<<BB * 512, 256, 0, stream>>>(feat, abuf);
  k_head<<<1, 256, 0, stream>>>(abuf, lin1_W, g6, b6, lin2_W, lin2_b, g7, b7,
                                lin3_W, lin3_b, out);
}
